// DD_a_90555090469568
// MI455X (gfx1250) — matrix operand order repaired and checked by reading, not compiled, not run on hardware
//
#include <hip/hip_runtime.h>
#include <hip/hip_bf16.h>
#include <math.h>

typedef __attribute__((ext_vector_type(16))) _Float16 v16h;
typedef __attribute__((ext_vector_type(8)))  _Float16 v8h;
typedef __attribute__((ext_vector_type(8)))  float    v8f;

#define NEG_SLOPE 0.2f

__device__ __forceinline__ float lrelu(float x) { return x > 0.f ? x : NEG_SLOPE * x; }

// Order-preserving float -> u32 key (monotonic), so atomicMax(u32) == float max.
__device__ __forceinline__ unsigned fkey(float f) {
    unsigned u = __float_as_uint(f);
    return (u & 0x80000000u) ? ~u : (u | 0x80000000u);
}
__device__ __forceinline__ float kinv(unsigned k) {
    unsigned u = (k & 0x80000000u) ? (k & 0x7FFFFFFFu) : ~k;
    return __uint_as_float(u);
}

// ---- A-fragment loaders (contiguous layout: 16 consecutive K from base)
__device__ __forceinline__ v16h load_a_frag(const float* __restrict__ arow) {
    v16h a;
#pragma unroll
    for (int e = 0; e < 8; ++e) a[e] = (_Float16)arow[e];
#pragma unroll
    for (int e = 0; e < 8; ++e) a[8 + e] = (_Float16)arow[8 + e];
    return a;
}
__device__ __forceinline__ v16h load_a_frag(const _Float16* __restrict__ arow) {
    v8h lo = *(const v8h*)arow;          // 16B aligned, one b128 load
    v8h hi = *(const v8h*)(arow + 8);    // 16B aligned, one b128 load
    return __builtin_shufflevector(lo, hi, 0, 1, 2, 3, 4, 5, 6, 7,
                                           8, 9, 10, 11, 12, 13, 14, 15);
}

// ---------------------------------------------------------------------------
// H[nrows x 64] = A[nrows x K] @ B[K x 64].  Bt is B pre-transposed to
// column-major f16 (Bt[col*K + k]) so every B fragment is 16 contiguous f16.
// One wave per 16-row tile; 4 f32 accumulators cover all 64 output columns.
// ---------------------------------------------------------------------------
template <typename AT>
__global__ void gemm_wmma(const AT* __restrict__ A, const _Float16* __restrict__ Bt,
                          float* __restrict__ H, int nrows, int K) {
    const int lane   = threadIdx.x & 31;
    const int wave   = threadIdx.x >> 5;
    const int tileM  = blockIdx.x * (blockDim.x >> 5) + wave;
    const int ntiles = (nrows + 15) >> 4;
    if (tileM >= ntiles) return;

    const int halfSel = lane >> 4;       // 0: lanes 0-15, 1: lanes 16-31
    const int laneLo  = lane & 15;
    const int row     = tileM * 16 + laneLo;
    const int rowLd   = row < nrows ? row : nrows - 1;
    const int kHalfA  = halfSel * 16;    // A fragment K-base within 32-wide k-tile
    const int kHalfB  = halfSel * 16;    // B fragment K-base within 32-wide k-tile

    v8f acc[4] = {};

    for (int kb = 0; kb < K; kb += 32) {
        const AT* arow = A + (size_t)rowLd * K + kb + kHalfA;
        if (kb + 32 < K) __builtin_prefetch(arow + 32, 0, 3);  // global_prefetch_b8
        v16h a = load_a_frag(arow);

#pragma unroll
        for (int nt = 0; nt < 4; ++nt) {
            // 16 consecutive K for this lane's column: one 32B (2x b128) load.
            const _Float16* bcol = Bt + (size_t)(nt * 16 + laneLo) * K + kb + kHalfB;
            v16h b = *(const v16h*)bcol;
            acc[nt] = __builtin_amdgcn_wmma_f32_16x16x32_f16(
                false, a, false, b, (short)0, acc[nt], false, false);
        }
    }

    // ---- store: VGPR v -> row v + 8*halfSel, col laneLo within nt tile ----
#pragma unroll
    for (int nt = 0; nt < 4; ++nt) {
#pragma unroll
        for (int v = 0; v < 8; ++v) {
            int r = tileM * 16 + v + 8 * halfSel;
            if (r < nrows) H[(size_t)r * 64 + nt * 16 + laneLo] = acc[nt][v];
        }
    }
}

// Convert W [K x 64] f32 row-major -> Wt [64 x K] f16 column-major.
__global__ void w_to_f16t(const float* __restrict__ W, _Float16* __restrict__ Wt, int K) {
    int i = blockIdx.x * blockDim.x + threadIdx.x;
    if (i >= K * 64) return;
    int k = i >> 6, c = i & 63;
    Wt[(size_t)c * K + k] = (_Float16)W[i];
}

// ---------------------------------------------------------------------------
// Per-node attention logits: s_src[i] = H[i,:]·a_src, s_dst[i] = H[i,:]·a_dst
// ---------------------------------------------------------------------------
__global__ void node_scores(const float* __restrict__ H, const float* __restrict__ as,
                            const float* __restrict__ ad, float* __restrict__ ss,
                            float* __restrict__ sd, int n) {
    int i = blockIdx.x * blockDim.x + threadIdx.x;
    if (i >= n) return;
    const float* h = H + (size_t)i * 64;
    float s0 = 0.f, s1 = 0.f;
#pragma unroll
    for (int c = 0; c < 64; c += 4) {
        float4 hv = *(const float4*)(h + c);
        float4 av = *(const float4*)(as + c);
        float4 dv = *(const float4*)(ad + c);
        s0 += hv.x * av.x + hv.y * av.y + hv.z * av.z + hv.w * av.w;
        s1 += hv.x * dv.x + hv.y * dv.y + hv.z * dv.z + hv.w * dv.w;
    }
    ss[i] = s0;
    sd[i] = s1;
}

// Zero accumulator [n*64], max-keys and denominators [n].
__global__ void init_accum(float* __restrict__ acc, unsigned* __restrict__ mkey,
                           float* __restrict__ denom, int n) {
    int i = blockIdx.x * blockDim.x + threadIdx.x;
    if (i < n * 64) acc[i] = 0.f;
    if (i < n) { mkey[i] = 0u; denom[i] = 0.f; }
}

// Pass 1: edge logit + segment max (via monotone u32 key atomicMax).
__global__ void edge_max_k(const long long* __restrict__ ei, int E, int Etot,
                           const float* __restrict__ ss, const float* __restrict__ sd,
                           float* __restrict__ alpha, unsigned* __restrict__ mkey) {
    int e = blockIdx.x * blockDim.x + threadIdx.x;
    if (e >= Etot) return;
    int s, d;
    if (e < E) { s = (int)ei[e]; d = (int)ei[(size_t)E + e]; }
    else       { s = e - E; d = s; }
    float a = lrelu(ss[s] + sd[d]);
    alpha[e] = a;
    atomicMax(mkey + d, fkey(a));
}

// Pass 2: exponentiate (in place) + segment sum of exps.
__global__ void edge_exp_k(const long long* __restrict__ ei, int E, int Etot,
                           float* __restrict__ alpha, const unsigned* __restrict__ mkey,
                           float* __restrict__ denom) {
    int e = blockIdx.x * blockDim.x + threadIdx.x;
    if (e >= Etot) return;
    int d = (e < E) ? (int)ei[(size_t)E + e] : (e - E);
    float ex = __expf(alpha[e] - kinv(mkey[d]));
    alpha[e] = ex;
    atomicAdd(denom + d, ex);
}

// Pass 3: out[dst,:] += coef * H[src,:].  16 threads per edge, float4 chunks:
// a wave's gathers cover two contiguous 256B feature rows.
__global__ void edge_agg_k(const long long* __restrict__ ei, int E, int Etot,
                           const float* __restrict__ ex, const float* __restrict__ denom,
                           const float* __restrict__ H, float* __restrict__ acc) {
    int t = blockIdx.x * blockDim.x + threadIdx.x;
    int e = t >> 4;
    if (e >= Etot) return;
    int c = (t & 15) * 4;
    int s, d;
    if (e < E) { s = (int)ei[e]; d = (int)ei[(size_t)E + e]; }
    else       { s = e - E; d = s; }
    float coef = ex[e] / denom[d];
    float4 hv = *(const float4*)(H + (size_t)s * 64 + c);
    float* o = acc + (size_t)d * 64 + c;
    atomicAdd(o + 0, coef * hv.x);
    atomicAdd(o + 1, coef * hv.y);
    atomicAdd(o + 2, coef * hv.z);
    atomicAdd(o + 3, coef * hv.w);
}

// Layer-1 epilogue: actH = leaky_relu(acc + b1) stored directly as f16
// (the f32 activation is consumed only by GEMM-2, which wants f16 anyway).
__global__ void finish_act_k(const float* __restrict__ acc, const float* __restrict__ b,
                             _Float16* __restrict__ actH, int total) {
    int i = blockIdx.x * blockDim.x + threadIdx.x;
    if (i >= total) return;
    actH[i] = (_Float16)lrelu(acc[i] + b[i & 63]);
}

// Layer-2 epilogue: out = acc + b2.
__global__ void finish_out_k(const float* __restrict__ acc, const float* __restrict__ b,
                             float* __restrict__ out, int total) {
    int i = blockIdx.x * blockDim.x + threadIdx.x;
    if (i >= total) return;
    out[i] = acc[i] + b[i & 63];
}

extern "C" void kernel_launch(void* const* d_in, const int* in_sizes, int n_in,
                              void* d_out, int out_size, void* d_ws, size_t ws_size,
                              hipStream_t stream) {
    const float*     x   = (const float*)d_in[0];
    const long long* ei  = (const long long*)d_in[1];
    const float*     W1  = (const float*)d_in[2];
    const float*     as1 = (const float*)d_in[3];
    const float*     ad1 = (const float*)d_in[4];
    const float*     b1  = (const float*)d_in[5];
    const float*     W2  = (const float*)d_in[6];
    const float*     as2 = (const float*)d_in[7];
    const float*     ad2 = (const float*)d_in[8];
    const float*     b2  = (const float*)d_in[9];

    const int N    = in_sizes[0] / 128;   // 100000
    const int E    = in_sizes[1] / 2;     // 1600000
    const int Etot = E + N;               // + self loops

    // Workspace carve-out (256B aligned).
    char*  ws  = (char*)d_ws;
    size_t off = 0;
    auto carve = [&](size_t bytes) {
        void* p = ws + off;
        off = (off + bytes + 255) & ~(size_t)255;
        return p;
    };
    float*     bufA  = (float*)carve((size_t)N * 64 * 4);      // H of current layer
    float*     bufB  = (float*)carve((size_t)N * 64 * 4);      // edge-softmax accumulator
    _Float16*  actH  = (_Float16*)carve((size_t)N * 64 * 2);   // layer-1 activation, f16
    _Float16*  W1t   = (_Float16*)carve((size_t)128 * 64 * 2); // W1^T f16 (col-major)
    _Float16*  W2t   = (_Float16*)carve((size_t)64 * 64 * 2);  // W2^T f16 (col-major)
    float*     ssrc  = (float*)carve((size_t)N * 4);
    float*     sdst  = (float*)carve((size_t)N * 4);
    float*     alpha = (float*)carve((size_t)Etot * 4);        // logit -> exp, in place
    unsigned*  mkey  = (unsigned*)carve((size_t)N * 4);
    float*     denom = (float*)carve((size_t)N * 4);

    const int BLK      = 256;
    const int ntilesM  = (N + 15) / 16;
    const int gemmGrid = (ntilesM + 3) / 4;                    // 4 waves / 128-thread block
    const int nodeGrid = (N + BLK - 1) / BLK;
    const int featGrid = (N * 64 + BLK - 1) / BLK;
    const int edgeGrid = (Etot + BLK - 1) / BLK;
    const int aggGrid  = (int)(((size_t)Etot * 16 + BLK - 1) / BLK);

    // Weight prep (tiny): f32 row-major -> f16 column-major.
    w_to_f16t<<<(128 * 64 + BLK - 1) / BLK, BLK, 0, stream>>>(W1, W1t, 128);
    w_to_f16t<<<(64 * 64 + BLK - 1) / BLK, BLK, 0, stream>>>(W2, W2t, 64);

    // ---------------- Layer 1 ----------------
    gemm_wmma<float><<<gemmGrid, 128, 0, stream>>>(x, W1t, bufA, N, 128);
    node_scores <<<nodeGrid, BLK, 0, stream>>>(bufA, as1, ad1, ssrc, sdst, N);
    init_accum  <<<featGrid, BLK, 0, stream>>>(bufB, mkey, denom, N);
    edge_max_k  <<<edgeGrid, BLK, 0, stream>>>(ei, E, Etot, ssrc, sdst, alpha, mkey);
    edge_exp_k  <<<edgeGrid, BLK, 0, stream>>>(ei, E, Etot, alpha, mkey, denom);
    edge_agg_k  <<<aggGrid,  BLK, 0, stream>>>(ei, E, Etot, alpha, denom, bufA, bufB);
    finish_act_k<<<featGrid, BLK, 0, stream>>>(bufB, b1, actH, N * 64);

    // ---------------- Layer 2 ----------------
    gemm_wmma<_Float16><<<gemmGrid, 128, 0, stream>>>(actH, W2t, bufA, N, 64);
    node_scores <<<nodeGrid, BLK, 0, stream>>>(bufA, as2, ad2, ssrc, sdst, N);
    init_accum  <<<featGrid, BLK, 0, stream>>>(bufB, mkey, denom, N);   // bufB -> acc
    edge_max_k  <<<edgeGrid, BLK, 0, stream>>>(ei, E, Etot, ssrc, sdst, alpha, mkey);
    edge_exp_k  <<<edgeGrid, BLK, 0, stream>>>(ei, E, Etot, alpha, mkey, denom);
    edge_agg_k  <<<aggGrid,  BLK, 0, stream>>>(ei, E, Etot, alpha, denom, bufA, bufB);
    finish_out_k<<<featGrid, BLK, 0, stream>>>(bufB, b2, (float*)d_out, N * 64);
}
